// TMK_46608985096827
// MI455X (gfx1250) — compile-verified, matrix operand order checked by reading, not run on hardware
//
#include <hip/hip_runtime.h>
#include <hip/hip_fp16.h>

// Shapes from the reference
#define N_ 8192
#define M_ 1024
#define D_ 16
#define KT_ (M_ / 32)   // 32 K-tiles of depth 32
#define NT_ (M_ / 16)   // 64 N-tiles of width 16

typedef __attribute__((ext_vector_type(16))) _Float16 v16h;
typedef __attribute__((ext_vector_type(8)))  _Float16 v8h;
typedef __attribute__((ext_vector_type(8)))  float    v8f;

// Assemble a 16-element f16 WMMA operand from two 16-byte chunks at p and p+gap.
__device__ __forceinline__ v16h ld_op(const _Float16* p, int gap) {
  v8h lo = *(const v8h*)p;
  v8h hi = *(const v8h*)(p + gap);
  return __builtin_shufflevector(lo, hi, 0, 1, 2, 3, 4, 5, 6, 7, 8, 9, 10, 11,
                                 12, 13, 14, 15);
}

// ---------------------------------------------------------------------------
// Kernel 1: LayerNorm + Laplace product kernel -> kmat[n,m] in f16 (row-major)
// One block per row n; 256 threads cover M=1024 columns (4 each).
// ---------------------------------------------------------------------------
__global__ __launch_bounds__(256) void k_ln_laplace(
    const float* __restrict__ x, const float* __restrict__ dp,
    _Float16* __restrict__ kmat) {
  const int n = blockIdx.x;
  const float* xr = x + (size_t)n * D_;

  float xv[D_];
  float mean = 0.f;
#pragma unroll
  for (int d = 0; d < D_; ++d) { xv[d] = xr[d]; mean += xv[d]; }
  mean *= (1.0f / D_);
  float var = 0.f;
#pragma unroll
  for (int d = 0; d < D_; ++d) { float t = xv[d] - mean; var += t * t; }
  const float rstd = rsqrtf(var * (1.0f / D_) + 1e-5f);
#pragma unroll
  for (int d = 0; d < D_; ++d) xv[d] = (xv[d] - mean) * rstd;

  const int m0 = threadIdx.x * 4;
#pragma unroll
  for (int i = 0; i < 4; ++i) {
    const int m = m0 + i;
    const float* dr = dp + (size_t)m * D_;
    float l1 = 0.f;
#pragma unroll
    for (int d = 0; d < D_; ++d) l1 += fabsf(xv[d] - dr[d]);
    kmat[(size_t)n * M_ + m] = (_Float16)__expf(-l1);  // lengthscale = 1
  }
}

// ---------------------------------------------------------------------------
// Kernel 2: prepack chol_inv (f32 [M,M], K-major rows) into WMMA B-operand
// order, f16. Tile (kt,nt) -> 512 f16 stored so lane L reads its 16 values
// from one contiguous 32-byte chunk.
//   lane<16 : N = nt*16+lane,    K = kt*32 + {0..7, 16..23}
//   lane>=16: N = nt*16+lane-16, K = kt*32 + {8..15, 24..31}
// ---------------------------------------------------------------------------
__global__ __launch_bounds__(256) void k_pack_b(
    const float* __restrict__ chol, _Float16* __restrict__ bp) {
  const int gid  = blockIdx.x * 256 + threadIdx.x;
  const int lane = gid & 31;
  const int tile = gid >> 5;              // tile = nt*KT_ + kt
  const int kt   = tile % KT_;
  const int nn   = (tile / KT_) * 16 + (lane & 15);
  const int kb   = kt * 32 + ((lane & 16) ? 8 : 0);

  _Float16* out = bp + (size_t)tile * 512 + lane * 16;
#pragma unroll
  for (int j = 0; j < 8; ++j)
    out[j] = (_Float16)chol[(size_t)(kb + j) * M_ + nn];
#pragma unroll
  for (int j = 0; j < 8; ++j)
    out[8 + j] = (_Float16)chol[(size_t)(kb + 16 + j) * M_ + nn];
}

// ---------------------------------------------------------------------------
// Kernel 3: GEMM out[N,M] = kmat(f16) @ Bpacked(f16), f32 accumulate.
// One wave32 computes a 16x64 output strip. The 8 waves of a block cover 8
// consecutive row-tiles of the SAME column group, so their B-panel loads are
// identical addresses -> WGP$ (L0) hits; B L2 traffic drops ~8x.
// K loop is software-pipelined: tile kt+1's 10 b128 loads are in flight while
// tile kt's 4 WMMAs execute (partial s_wait_loadcnt instead of 0).
// ---------------------------------------------------------------------------
__global__ __launch_bounds__(256) void k_gemm(
    const _Float16* __restrict__ kmat, const _Float16* __restrict__ bp,
    float* __restrict__ out) {
  const int wave    = (blockIdx.x * 256 + threadIdx.x) >> 5;
  const int lane    = threadIdx.x & 31;
  const int tileRow = wave & (N_ / 16 - 1);   // 512 row-tiles, consecutive in block
  const int tileCol = wave >> 9;              // 16 column groups
  const int n0 = tileRow * 16;
  const int m0 = tileCol * 64;

  // Per-lane A base: row n0 + lane%16, K offset 0 (lanes 0-15) or 8 (16-31)
  const _Float16* arow =
      kmat + (size_t)(n0 + (lane & 15)) * M_ + ((lane & 16) ? 8 : 0);
  // Per-lane B base for column group tiles j=0..3 at kt: +j*KT_*512
  const _Float16* brow = bp + (size_t)(m0 / 16) * KT_ * 512 + lane * 16;

  v8f c0 = {}, c1 = {}, c2 = {}, c3 = {};

  // ---- prologue: load tile kt=0 ----
  v16h A  = ld_op(arow, 16);
  v16h B0 = ld_op(brow + 0 * (KT_ * 512), 8);
  v16h B1 = ld_op(brow + 1 * (KT_ * 512), 8);
  v16h B2 = ld_op(brow + 2 * (KT_ * 512), 8);
  v16h B3 = ld_op(brow + 3 * (KT_ * 512), 8);

  for (int kt = 0; kt < KT_ - 1; ++kt) {
    // issue next tile's loads first (they overlap the WMMAs below)
    const _Float16* apn = arow + (kt + 1) * 32;
    const _Float16* bpn = brow + (size_t)(kt + 1) * 512;
    __builtin_prefetch(apn + 64, 0, 1);  // A two tiles ahead -> global_prefetch
    v16h An  = ld_op(apn, 16);
    v16h Bn0 = ld_op(bpn + 0 * (KT_ * 512), 8);
    v16h Bn1 = ld_op(bpn + 1 * (KT_ * 512), 8);
    v16h Bn2 = ld_op(bpn + 2 * (KT_ * 512), 8);
    v16h Bn3 = ld_op(bpn + 3 * (KT_ * 512), 8);

    c0 = __builtin_amdgcn_wmma_f32_16x16x32_f16(false, A, false, B0, (short)0, c0, false, false);
    c1 = __builtin_amdgcn_wmma_f32_16x16x32_f16(false, A, false, B1, (short)0, c1, false, false);
    c2 = __builtin_amdgcn_wmma_f32_16x16x32_f16(false, A, false, B2, (short)0, c2, false, false);
    c3 = __builtin_amdgcn_wmma_f32_16x16x32_f16(false, A, false, B3, (short)0, c3, false, false);

    A = An; B0 = Bn0; B1 = Bn1; B2 = Bn2; B3 = Bn3;
  }
  // ---- epilogue: last tile ----
  c0 = __builtin_amdgcn_wmma_f32_16x16x32_f16(false, A, false, B0, (short)0, c0, false, false);
  c1 = __builtin_amdgcn_wmma_f32_16x16x32_f16(false, A, false, B1, (short)0, c1, false, false);
  c2 = __builtin_amdgcn_wmma_f32_16x16x32_f16(false, A, false, B2, (short)0, c2, false, false);
  c3 = __builtin_amdgcn_wmma_f32_16x16x32_f16(false, A, false, B3, (short)0, c3, false, false);

  // C/D layout: lane<16 -> rows n0+v, lane>=16 -> rows n0+8+v; col = lane%16
  const int rowBase = n0 + ((lane & 16) ? 8 : 0);
  const int col     = (lane & 15);
  float* ob = out + (size_t)rowBase * M_ + m0 + col;
#pragma unroll
  for (int v = 0; v < 8; ++v) {
    ob[(size_t)v * M_ +  0] = c0[v];
    ob[(size_t)v * M_ + 16] = c1[v];
    ob[(size_t)v * M_ + 32] = c2[v];
    ob[(size_t)v * M_ + 48] = c3[v];
  }
}

// ---------------------------------------------------------------------------
extern "C" void kernel_launch(void* const* d_in, const int* in_sizes, int n_in,
                              void* d_out, int out_size, void* d_ws,
                              size_t ws_size, hipStream_t stream) {
  const float* x    = (const float*)d_in[0];  // [N, D]
  const float* dp   = (const float*)d_in[1];  // [M, D]
  const float* chol = (const float*)d_in[2];  // [M, M]
  float* out        = (float*)d_out;          // [N, M]

  _Float16* kmat  = (_Float16*)d_ws;                                // 16 MB
  _Float16* bpack = (_Float16*)((char*)d_ws + (size_t)N_ * M_ * 2); // 2 MB

  // 1) LayerNorm + Laplace kernel matrix (f16)
  k_ln_laplace<<<N_, 256, 0, stream>>>(x, dp, kmat);

  // 2) Pack chol_inv into WMMA B-operand layout (f16). KT_*NT_*32 lanes.
  k_pack_b<<<(KT_ * NT_ * 32) / 256, 256, 0, stream>>>(chol, bpack);

  // 3) WMMA GEMM: (N/16)*(M/64) waves, 8 waves per block.
  k_gemm<<<(N_ / 16) * (M_ / 64) / 8, 256, 0, stream>>>(kmat, bpack, out);
}
